// MultiHeadSelfAttention_34608846471467
// MI455X (gfx1250) — matrix operand order repaired and checked by reading, not compiled, not run on hardware
//
#include <hip/hip_runtime.h>
#include <hip/hip_bf16.h>
#include <math.h>

// ---------------------------------------------------------------------------
// MI455X (gfx1250) wave32 WMMA implementation of causal MHA block.
// All matmuls (QKV proj, QK^T, PV, O proj) use v_wmma_f32_16x16x32_bf16.
// V is pre-transposed to [b,h][dh,seq] so every WMMA operand is loaded with
// contiguous 128-bit vector loads.
// ---------------------------------------------------------------------------

typedef __bf16 bf16_t;
typedef __attribute__((ext_vector_type(16))) __bf16 v16bf;
typedef __attribute__((ext_vector_type(8)))  float  v8f;

union Frag16 {
    v16bf v;
    uint4 q[2];
    unsigned short u[16];
    __bf16 h[16];
};

static __device__ inline v8f wmma_bf16(v16bf a, v16bf b, v8f c) {
    return __builtin_amdgcn_wmma_f32_16x16x32_bf16(
        /*neg_a=*/false, a, /*neg_b=*/false, b,
        /*c_mod=*/(short)0, c, /*reuse_a=*/false, /*reuse_b=*/false);
}

// A-matrix fragment (16x32 bf16): lane holds row M=lane&15.
// j=0..15 -> K = 16*half + j (16 contiguous elements).
// Caller passes p = &A[(row0 + l16)*ld + k0 + 16*half]  (16B aligned).
static __device__ inline v16bf load_a16(const bf16_t* p) {
    Frag16 f;
    f.q[0] = *(const uint4*)(p);
    f.q[1] = *(const uint4*)(p + 8);
    return f.v;
}

// B-matrix fragment (32x16 bf16): lane holds col N=lane&15.
// j=0..15 -> K = 16*half + j (16 contiguous elements).
// Caller passes p = &B^T[(n0 + l16)*ld + k0 + 16*half]  (16B aligned).
static __device__ inline v16bf load_b16(const bf16_t* p) {
    Frag16 f;
    f.q[0] = *(const uint4*)(p);
    f.q[1] = *(const uint4*)(p + 8);
    return f.v;
}

// Row reductions across the 16-lane groups that share a C-matrix row.
static __device__ inline float rowmax16(float x) {
    #pragma unroll
    for (int m = 1; m < 16; m <<= 1) x = fmaxf(x, __shfl_xor(x, m, 32));
    return x;
}
static __device__ inline float rowsum16(float x) {
    #pragma unroll
    for (int m = 1; m < 16; m <<= 1) x += __shfl_xor(x, m, 32);
    return x;
}

#define DIM 512
#define SEQ 4096
#define NHEAD 8
#define DHEAD 64

// --------------------------- weight convert + transpose --------------------
__global__ void mha_cvtw_kernel(const float* __restrict__ W, bf16_t* __restrict__ Wt) {
    int i = blockIdx.x * blockDim.x + threadIdx.x;   // over DIM*DIM
    int k = i >> 9;          // row in  [K, N]
    int n = i & (DIM - 1);   // col
    Wt[(size_t)n * DIM + k] = (bf16_t)W[i];
}

// --------------------------- V transpose to [bh][dh][seq] ------------------
__global__ void mha_vtrans_kernel(const bf16_t* __restrict__ V, bf16_t* __restrict__ Vt) {
    int i = blockIdx.x * blockDim.x + threadIdx.x;   // over R*DIM
    int row = i >> 9;                                 // b*SEQ + s
    int col = i & (DIM - 1);                          // h*64 + d
    int b = row >> 12, s = row & (SEQ - 1);
    int h = col >> 6,  d = col & (DHEAD - 1);
    Vt[(((size_t)(b * NHEAD + h) * DHEAD) + d) * SEQ + s] = V[i];
}

// --------------------------- RMSNorm ---------------------------------------
__global__ void mha_rmsnorm_kernel(const float* __restrict__ x,
                                   const float* __restrict__ scale,
                                   bf16_t* __restrict__ xn) {
    __shared__ float red[256];
    int row = blockIdx.x;
    const float* xr = x + (size_t)row * DIM;
    float s = 0.f;
    for (int c = threadIdx.x; c < DIM; c += 256) { float v = xr[c]; s += v * v; }
    red[threadIdx.x] = s;
    __syncthreads();
    #pragma unroll
    for (int off = 128; off > 0; off >>= 1) {
        if (threadIdx.x < off) red[threadIdx.x] += red[threadIdx.x + off];
        __syncthreads();
    }
    float rr = rsqrtf(red[0] * (1.0f / DIM) + 1e-6f);
    for (int c = threadIdx.x; c < DIM; c += 256)
        xn[(size_t)row * DIM + c] = (bf16_t)(xr[c] * rr * scale[c]);
}

// --------------------------- generic bf16 WMMA GEMM ------------------------
// C[M,N] = A[M,K] * B[K,N], B given pre-transposed (Bt[N,K]).
// Block = 8 waves; each wave computes one 16x16 tile; block covers 16x128.
__global__ void mha_gemm_kernel(const bf16_t* __restrict__ A,
                                const bf16_t* __restrict__ Bt,
                                float* __restrict__ Cf,
                                bf16_t* __restrict__ Cb,
                                int M, int N, int K, int outF32) {
    int lane = threadIdx.x & 31;
    int w    = threadIdx.x >> 5;
    int l16  = lane & 15, half = lane >> 4;
    int row0 = blockIdx.x * 16;
    int n0   = blockIdx.y * 128 + w * 16;

    const bf16_t* ap = A  + (size_t)(row0 + l16) * K + 16 * half;
    const bf16_t* bp = Bt + (size_t)(n0   + l16) * K + 16 * half;

    v8f acc = {};
    for (int k0 = 0; k0 < K; k0 += 32) {
        __builtin_prefetch(bp + k0 + 128, 0, 0);     // global_prefetch_b8
        v16bf a = load_a16(ap + k0);
        v16bf b = load_b16(bp + k0);
        acc = wmma_bf16(a, b, acc);
    }
    #pragma unroll
    for (int r = 0; r < 8; r++) {
        size_t m = (size_t)(row0 + r + 8 * half);
        size_t n = (size_t)(n0 + l16);
        if (outF32) Cf[m * N + n] = acc[r];
        else        Cb[m * N + n] = (bf16_t)acc[r];
    }
}

// --------------------------- RoPE (in-place on Q and K) --------------------
__global__ void mha_rope_kernel(bf16_t* __restrict__ Q, bf16_t* __restrict__ Kb) {
    const int halfd = DHEAD / 2;                      // 32
    int idx = blockIdx.x * blockDim.x + threadIdx.x;  // rows * NHEAD * halfd
    int i   = idx & (halfd - 1);
    int hd  = (idx >> 5) & (NHEAD - 1);
    size_t row = (size_t)(idx >> 8);                  // / (32*8)
    int pos = (int)(row & (SEQ - 1));
    float theta = __powf(1.0e6f, -(float)i / (float)halfd);
    float ang = (float)pos * theta;
    float sn = __sinf(ang), cs = __cosf(ang);
    size_t base = row * DIM + hd * DHEAD;

    float q1 = (float)Q[base + i], q2 = (float)Q[base + halfd + i];
    Q[base + i]         = (bf16_t)(q1 * cs - q2 * sn);
    Q[base + halfd + i] = (bf16_t)(q1 * sn + q2 * cs);
    float k1 = (float)Kb[base + i], k2 = (float)Kb[base + halfd + i];
    Kb[base + i]         = (bf16_t)(k1 * cs - k2 * sn);
    Kb[base + halfd + i] = (bf16_t)(k1 * sn + k2 * cs);
}

// --------------------------- flash attention (causal) ----------------------
// grid.x = b*NHEAD, grid.y = SEQ/128. 8 waves/block; wave owns a 16-query tile.
// K is row-major [row, DIM] (head slice); V is pre-transposed [bh][dh][seq].
__global__ void mha_attn_kernel(const bf16_t* __restrict__ Q,
                                const bf16_t* __restrict__ Km,
                                const bf16_t* __restrict__ Vt,
                                bf16_t* __restrict__ O) {
    int lane = threadIdx.x & 31;
    int w    = threadIdx.x >> 5;
    int l16  = lane & 15, half = lane >> 4;
    int bh   = blockIdx.x;
    int b    = bh >> 3, hd = bh & 7;
    int q0   = blockIdx.y * 128 + w * 16;
    size_t rowBase = (size_t)b * SEQ;
    int hoff = hd * DHEAD;
    const bf16_t* vbase = Vt + (size_t)bh * DHEAD * SEQ;

    __shared__ unsigned short pbuf[8][16 * 32];
    unsigned short* pl = pbuf[w];

    // Q fragments for both dh-chunks (dh=64 -> K chunks 0 and 32), loaded once.
    const bf16_t* qp = Q + (rowBase + q0 + l16) * DIM + hoff + 16 * half;
    v16bf aq0 = load_a16(qp);
    v16bf aq1 = load_a16(qp + 32);

    v8f oacc[4] = {};
    float mrow[8], lrow[8];
    #pragma unroll
    for (int r = 0; r < 8; r++) { mrow[r] = -__builtin_inff(); lrow[r] = 0.f; }
    const float rscale = 0.125f;                      // 1/sqrt(64)

    int nkb = (q0 + 16 + 31) >> 5;                    // causal: keys <= q0+15
    for (int kb = 0; kb < nkb; kb++) {
        int s0 = kb * 32;
        // ---- S = Q K^T for 32 keys (two 16-key N tiles, two dh chunks each)
        v8f c0 = {}, c1 = {};
        {
            const bf16_t* kp0 = Km + (rowBase + s0 + l16) * DIM + hoff + 16 * half;
            c0 = wmma_bf16(aq0, load_b16(kp0), c0);
            c0 = wmma_bf16(aq1, load_b16(kp0 + 32), c0);
            const bf16_t* kp1 = Km + (rowBase + s0 + 16 + l16) * DIM + hoff + 16 * half;
            c1 = wmma_bf16(aq0, load_b16(kp1), c1);
            c1 = wmma_bf16(aq1, load_b16(kp1 + 32), c1);
        }
        // ---- online softmax update (row = r + 8*half, key col = l16)
        float p0[8], p1[8];
        #pragma unroll
        for (int r = 0; r < 8; r++) {
            int t = q0 + r + 8 * half;
            float s0v = c0[r] * rscale; if (s0 + l16 > t)      s0v = -__builtin_inff();
            float s1v = c1[r] * rscale; if (s0 + 16 + l16 > t) s1v = -__builtin_inff();
            float rm = rowmax16(fmaxf(s0v, s1v));
            float mn = fmaxf(mrow[r], rm);
            float alpha = __expf(mrow[r] - mn);
            p0[r] = __expf(s0v - mn);
            p1[r] = __expf(s1v - mn);
            float rs = rowsum16(p0[r] + p1[r]);
            lrow[r] = lrow[r] * alpha + rs;
            mrow[r] = mn;
            oacc[0][r] *= alpha; oacc[1][r] *= alpha;
            oacc[2][r] *= alpha; oacc[3][r] *= alpha;
        }
        // ---- transpose P (C layout) -> A layout through per-wave LDS
        #pragma unroll
        for (int r = 0; r < 8; r++) {
            int rowm = r + 8 * half;
            __bf16 h0 = (bf16_t)p0[r], h1 = (bf16_t)p1[r];
            pl[rowm * 32 + l16]      = __builtin_bit_cast(unsigned short, h0);
            pl[rowm * 32 + 16 + l16] = __builtin_bit_cast(unsigned short, h1);
        }
        asm volatile("s_wait_dscnt 0" ::: "memory");
        Frag16 fp;
        const unsigned short* pr = pl + l16 * 32;
        *(uint4*)&fp.u[0] = *(const uint4*)(pr + 16 * half);
        *(uint4*)&fp.u[8] = *(const uint4*)(pr + 16 * half + 8);
        // ---- O += P V : inner dim = 32 keys, 4 N tiles of dh.
        // V^T rows are contiguous in seq -> 2x b128 loads per fragment.
        #pragma unroll
        for (int n = 0; n < 4; n++) {
            v16bf fv = load_b16(vbase + (size_t)(n * 16 + l16) * SEQ + s0 + 16 * half);
            oacc[n] = wmma_bf16(fp.v, fv, oacc[n]);
        }
    }
    // ---- epilogue: O / l -> bf16 attention output [row, DIM]
    #pragma unroll
    for (int r = 0; r < 8; r++) {
        float inv = 1.0f / lrow[r];
        size_t orow = (rowBase + q0 + r + 8 * half) * DIM + hoff;
        #pragma unroll
        for (int n = 0; n < 4; n++)
            O[orow + n * 16 + l16] = (bf16_t)(oacc[n][r] * inv);
    }
}

// ---------------------------------------------------------------------------
extern "C" void kernel_launch(void* const* d_in, const int* in_sizes, int n_in,
                              void* d_out, int out_size, void* d_ws, size_t ws_size,
                              hipStream_t stream) {
    (void)in_sizes; (void)n_in; (void)out_size; (void)ws_size;
    const float* x     = (const float*)d_in[0];
    const float* scale = (const float*)d_in[1];
    const float* Wq    = (const float*)d_in[2];
    const float* Wk    = (const float*)d_in[3];
    const float* Wv    = (const float*)d_in[4];
    const float* Wo    = (const float*)d_in[5];

    const int B = 2, R = B * SEQ;                     // 8192 rows
    bf16_t* xn  = (bf16_t*)d_ws;                      // R*DIM
    bf16_t* wqT = xn  + (size_t)R * DIM;
    bf16_t* wkT = wqT + (size_t)DIM * DIM;
    bf16_t* wvT = wkT + (size_t)DIM * DIM;
    bf16_t* woT = wvT + (size_t)DIM * DIM;
    bf16_t* Qb  = woT + (size_t)DIM * DIM;
    bf16_t* Kb  = Qb  + (size_t)R * DIM;
    bf16_t* Vb  = Kb  + (size_t)R * DIM;
    bf16_t* Vtb = Vb  + (size_t)R * DIM;              // [B*NHEAD][DHEAD][SEQ]
    bf16_t* AO  = Vtb + (size_t)R * DIM;

    const int wgrid = (DIM * DIM) / 256;
    mha_cvtw_kernel<<<wgrid, 256, 0, stream>>>(Wq, wqT);
    mha_cvtw_kernel<<<wgrid, 256, 0, stream>>>(Wk, wkT);
    mha_cvtw_kernel<<<wgrid, 256, 0, stream>>>(Wv, wvT);
    mha_cvtw_kernel<<<wgrid, 256, 0, stream>>>(Wo, woT);

    mha_rmsnorm_kernel<<<R, 256, 0, stream>>>(x, scale, xn);

    dim3 gg(R / 16, DIM / 128);
    mha_gemm_kernel<<<gg, 256, 0, stream>>>(xn, wqT, nullptr, Qb, R, DIM, DIM, 0);
    mha_gemm_kernel<<<gg, 256, 0, stream>>>(xn, wkT, nullptr, Kb, R, DIM, DIM, 0);
    mha_gemm_kernel<<<gg, 256, 0, stream>>>(xn, wvT, nullptr, Vb, R, DIM, DIM, 0);

    mha_rope_kernel<<<(R * NHEAD * (DHEAD / 2)) / 256, 256, 0, stream>>>(Qb, Kb);

    mha_vtrans_kernel<<<(R * DIM) / 256, 256, 0, stream>>>(Vb, Vtb);

    mha_attn_kernel<<<dim3(B * NHEAD, SEQ / 128), 256, 0, stream>>>(Qb, Kb, Vtb, AO);

    mha_gemm_kernel<<<gg, 256, 0, stream>>>(AO, woT, (float*)d_out, nullptr, R, DIM, DIM, 1);
}
